// DynamicViewSelection_66288525246506
// MI455X (gfx1250) — compile-verified, matrix operand order checked by reading, not run on hardware
//
#include <hip/hip_runtime.h>

// ---------------------------------------------------------------------------
// DynamicViewSelection for MI455X (gfx1250, wave32)
//
// Outputs (concatenated flat in d_out, float32):
//   [0      .. 2047]       selected_features (4 x 512)
//   [2048   .. 2051]       indices (4,) stored as float
//   [2052   .. 2052+154140672) rendered (1024 x 224 x 224 x 3)
//
// point_cloud input is dead code (only feeds discarded camera positions).
// Roofline: render stores = 616 MB -> ~26.5 us floor at 23.3 TB/s; everything
// else (0.27 GFLOP MLP, cosine, top-4) is noise. WMMA + async-to-LDS used for
// the MLP GEMM; NT b128 stores for the streaming render output.
// ---------------------------------------------------------------------------

typedef __attribute__((ext_vector_type(16))) _Float16 h16;
typedef __attribute__((ext_vector_type(8)))  _Float16 h8;
typedef __attribute__((ext_vector_type(8)))  float    v8f;
typedef __attribute__((ext_vector_type(4)))  float    v4f;

#define NCAND   1024
#define DFEAT   512
#define DHID    256
#define IMG     224
#define IMG_FLT (IMG*IMG*3)      // 150528 floats per candidate image
#define NVIEWS  4

#ifndef __has_builtin
#define __has_builtin(x) 0
#endif

// gfx1250 async global->LDS copy (ASYNCcnt-tracked, bypasses VGPRs).
#if __has_builtin(__builtin_amdgcn_global_load_async_to_lds_b128)
#define DVS_ASYNC_LDS 1
#else
#define DVS_ASYNC_LDS 0
#endif

#if DVS_ASYNC_LDS
// Exact pointee type per the compiler diagnostic:
//   '__attribute__((__vector_size__(4 * sizeof(int)))) int __device__ *'
typedef int dvs_v4i __attribute__((__vector_size__(4 * sizeof(int))));
typedef __attribute__((address_space(1))) dvs_v4i* dvs_gp;   // global (AS1)
typedef __attribute__((address_space(3))) dvs_v4i* dvs_lp;   // LDS    (AS3)

__device__ __forceinline__ void dvs_async_b128(const void* g, void* l)
{
    __builtin_amdgcn_global_load_async_to_lds_b128((dvs_gp)g, (dvs_lp)l, 0, 0);
}
__device__ __forceinline__ void dvs_wait_async0()
{
#if __has_builtin(__builtin_amdgcn_s_wait_asynccnt)
    __builtin_amdgcn_s_wait_asynccnt(0);
#else
    asm volatile("s_wait_asynccnt 0" ::: "memory");
#endif
}
#endif

// ---------------------------------------------------------------------------
// Kernel 1: cosine(view_features[row], text)  (f32, precision-sensitive)
// one block (64 threads) per row
// ---------------------------------------------------------------------------
__global__ __launch_bounds__(64)
void dvs_relevance_kernel(const float* __restrict__ vf,
                          const float* __restrict__ txt,
                          float* __restrict__ tr)
{
    __shared__ float s_dot[64], s_sq[64], s_tsq[64];
    const int row = blockIdx.x;
    const int t   = threadIdx.x;

    float dot = 0.f, sq = 0.f, tsq = 0.f;
    for (int c = t; c < DFEAT; c += 64) {
        float v  = vf[row * DFEAT + c];
        float tv = txt[c];
        dot += v * tv;
        sq  += v * v;
        tsq += tv * tv;
    }
    s_dot[t] = dot; s_sq[t] = sq; s_tsq[t] = tsq;
    __syncthreads();
    for (int off = 32; off > 0; off >>= 1) {
        if (t < off) {
            s_dot[t] += s_dot[t + off];
            s_sq[t]  += s_sq[t + off];
            s_tsq[t] += s_tsq[t + off];
        }
        __syncthreads();
    }
    if (t == 0) {
        float denom = fmaxf(sqrtf(s_sq[0]) * sqrtf(s_tsq[0]), 1e-8f);
        tr[row] = s_dot[0] / denom;
    }
}

// ---------------------------------------------------------------------------
// Kernel 2: coverage = sigmoid( relu(VF @ W1 + b1) @ w2 + b2 )
// WMMA f16->f32 GEMM. One block = one 16-row M tile. 4 waves; each wave owns
// four 16x16 N-tiles (4*4*16 = 256 = DHID). K-loop: 16 steps of K=32.
// Tiles staged via GLOBAL_LOAD_ASYNC_TO_LDS_B128 (f32), converted LDS->LDS to
// f16 (B transposed), then fed to v_wmma via two ds_load_b128 per fragment.
// Epilogue fuses bias+relu+dot(w2) via LDS float atomics (ds_add_f32).
// ---------------------------------------------------------------------------
__global__ __launch_bounds__(128)
void dvs_score_kernel(const float* __restrict__ vf,
                      const float* __restrict__ w1,
                      const float* __restrict__ b1,
                      const float* __restrict__ w2,
                      const float* __restrict__ b2,
                      float* __restrict__ cov)
{
    __shared__ __align__(16) _Float16 sA[16][32];   // f16 A tile (1 KB)
    __shared__ __align__(16) _Float16 sB[256][32];  // f16 B tile TRANSPOSED [n][k] (16 KB)
#if DVS_ASYNC_LDS
    __shared__ __align__(16) float    sAf[16][32];  // raw f32 A tile (2 KB)
    __shared__ __align__(16) float    sBf[32][256]; // raw f32 B tile [k][n] (32 KB)
#endif
    __shared__ float rowsum[16];

    const int tid  = threadIdx.x;
    const int wave = tid >> 5;
    const int lane = tid & 31;
    const int m0   = blockIdx.x * 16;

    if (tid < 16) rowsum[tid] = 0.0f;

    v8f acc[4] = {};                    // one f32 16x16 accumulator per N-tile

    const int g  = lane >> 4;           // lane half (selects K sub-ranges)
    const int ml = lane & 15;           // A: row within tile / B,D: col within tile

    for (int ks = 0; ks < DFEAT / 32; ++ks) {
        const int k0 = ks * 32;
        __syncthreads();                // LDS reuse fence (incl. rowsum init)

#if DVS_ASYNC_LDS
        // ---- async stage: raw f32 tiles straight into LDS ----
        {   // A: 512 floats -> one b128 per thread
            int e = tid * 4;
            int r = e >> 5, c = e & 31;
            dvs_async_b128(&vf[(m0 + r) * DFEAT + k0 + c], &sAf[r][c]);
        }
        #pragma unroll
        for (int j = 0; j < 16; ++j) {  // B: 8192 floats -> 16 b128 per thread
            int e = (tid + j * 128) * 4;
            int k = e >> 8, n = e & 255;
            dvs_async_b128(&w1[(k0 + k) * DHID + n], &sBf[k][n]);
        }
        dvs_wait_async0();
        __syncthreads();
        // ---- convert LDS f32 -> LDS f16 (B transposed for contiguous-K frags) ----
        for (int i = tid; i < 16 * 32; i += 128) {
            int r = i >> 5, c = i & 31;
            sA[r][c] = (_Float16)sAf[r][c];
        }
        for (int i = tid; i < 32 * 256; i += 128) {
            int k = i >> 8, n = i & 255;
            sB[n][k] = (_Float16)sBf[k][n];
        }
#else
        // ---- fallback: vectorized synchronous staging ----
        {   // A: one float4 per thread
            int e = tid * 4;
            int r = e >> 5, c = e & 31;
            v4f fv = *(const v4f*)&vf[(m0 + r) * DFEAT + k0 + c];
            #pragma unroll
            for (int q = 0; q < 4; ++q) sA[r][c + q] = (_Float16)fv[q];
        }
        #pragma unroll
        for (int j = 0; j < 16; ++j) {  // B: 16 float4 per thread, transposed store
            int e = (tid + j * 128) * 4;
            int k = e >> 8, n = e & 255;
            v4f fv = *(const v4f*)&w1[(k0 + k) * DHID + n];
            #pragma unroll
            for (int q = 0; q < 4; ++q) sB[n + q][k] = (_Float16)fv[q];
        }
#endif
        __syncthreads();

        // A fragment, per ISA 16-bit 16x32 layout:
        //   lanes 0-15:  halves0-7 = K0-7,  halves8-15 = K16-23
        //   lanes 16-31: halves0-7 = K8-15, halves8-15 = K24-31
        h8 a_lo = *(const h8*)&sA[ml][8 * g];
        h8 a_hi = *(const h8*)&sA[ml][16 + 8 * g];
        h16 a = __builtin_shufflevector(a_lo, a_hi,
                                        0,1,2,3,4,5,6,7,8,9,10,11,12,13,14,15);

        #pragma unroll
        for (int nt = 0; nt < 4; ++nt) {
            const int n = wave * 64 + nt * 16 + ml;
            // B stored column-major in LDS -> same contiguous-K pattern as A
            h8 b_lo = *(const h8*)&sB[n][8 * g];
            h8 b_hi = *(const h8*)&sB[n][16 + 8 * g];
            h16 b = __builtin_shufflevector(b_lo, b_hi,
                                            0,1,2,3,4,5,6,7,8,9,10,11,12,13,14,15);
            // (neg_a, A, neg_b, B, c_mod, C, reuse_a, reuse_b)
            acc[nt] = __builtin_amdgcn_wmma_f32_16x16x32_f16(
                          false, a, false, b, (short)0, acc[nt], false, false);
        }
    }
    __syncthreads();

    // Epilogue. D layout: lane(0-15) -> N=lane, VGPR r -> M=r; lanes 16-31 M=r+8.
    #pragma unroll
    for (int nt = 0; nt < 4; ++nt) {
        const int n  = wave * 64 + nt * 16 + ml;
        const float b1n = b1[n];
        const float w2n = w2[n];
        #pragma unroll
        for (int r = 0; r < 8; ++r) {
            float h = acc[nt][r] + b1n;
            h = fmaxf(h, 0.0f);
            atomicAdd(&rowsum[r + 8 * g], h * w2n);   // ds_add_f32
        }
    }
    __syncthreads();

    if (tid < 16) {
        float x = rowsum[tid] + b2[0];
        cov[m0 + tid] = 1.0f / (1.0f + __expf(-x));
    }
}

// ---------------------------------------------------------------------------
// Kernel 3: utility + top-4 (JAX tie-break: lower index wins) + gather
// single block of 1024 threads
// ---------------------------------------------------------------------------
__global__ __launch_bounds__(1024)
void dvs_select_kernel(const float* __restrict__ vf,
                       const float* __restrict__ tr,
                       const float* __restrict__ cov,
                       const float* __restrict__ wt_p,
                       const float* __restrict__ wc_p,
                       const float* __restrict__ wclip_p,
                       float* __restrict__ out)
{
    __shared__ float vals[NCAND];
    __shared__ float rv[NCAND];
    __shared__ int   ri[NCAND];
    __shared__ int   sel[NVIEWS];

    const int t = threadIdx.x;
    const float wt = *wt_p, wc = *wc_p, wclip = *wclip_p;
    const float wcn    = wc    / (wc + wclip);
    const float wclipn = wclip / (wc + wclip);

    const float trv = tr[t];
    vals[t] = wt * trv + wcn * cov[t] + wclipn * (0.8f * trv);
    __syncthreads();

    for (int k = 0; k < NVIEWS; ++k) {
        rv[t] = vals[t];
        ri[t] = t;
        __syncthreads();
        for (int off = 512; off > 0; off >>= 1) {
            if (t < off) {
                float v2 = rv[t + off];
                int   i2 = ri[t + off];
                if (v2 > rv[t] || (v2 == rv[t] && i2 < ri[t])) {
                    rv[t] = v2; ri[t] = i2;
                }
            }
            __syncthreads();
        }
        if (t == 0) { sel[k] = ri[0]; vals[ri[0]] = -3.0e38f; }
        __syncthreads();
    }

    if (t < NVIEWS) out[NVIEWS * DFEAT + t] = (float)sel[t];
    for (int i = t; i < NVIEWS * DFEAT; i += 1024) {
        int r = i >> 9, c = i & (DFEAT - 1);
        out[i] = vf[sel[r] * DFEAT + c];
    }
}

// ---------------------------------------------------------------------------
// Kernel 4: rendered images — the bandwidth-bound part (616 MB of stores).
// Each thread owns 4 consecutive floats of one image (16B-aligned), computes
// the radial intensity once, then streams it to a slice of 64 candidates with
// non-temporal (TH=NT) 128-bit stores: write-once data >> 192MB L2.
// grid = (150528/4/256, 16) = (147, 16)
// ---------------------------------------------------------------------------
__global__ __launch_bounds__(256)
void dvs_render_kernel(float* __restrict__ out)   // points at d_out + 2052
{
    const int vec = blockIdx.x * 256 + threadIdx.x;   // 0 .. 37631
    const int f0  = vec * 4;
    const int p0  = f0 / 3;

    float inten[2];
    #pragma unroll
    for (int pi = 0; pi < 2; ++pi) {
        int p = p0 + pi;
        if (p > IMG * IMG - 1) p = IMG * IMG - 1;
        int y = p / IMG;
        int x = p - y * IMG;
        float cy = (float)(y - 112) * (1.0f / 112.0f);
        float cx = (float)(x - 112) * (1.0f / 112.0f);
        float d  = sqrtf(cy * cy + cx * cx);
        inten[pi] = __expf(-2.0f * d);
    }

    const float chan[3] = {1.0f, 0.8f, 0.6f};
    float v[4];
    #pragma unroll
    for (int j = 0; j < 4; ++j) {
        int f  = f0 + j;
        int p  = f / 3;
        int ch = f - p * 3;
        v[j] = inten[p - p0] * chan[ch];
    }
    v4f val = { v[0], v[1], v[2], v[3] };

    const int c0 = blockIdx.y * 64;                   // 16 slices x 64 candidates
    float* base = out + (size_t)c0 * IMG_FLT + f0;
    for (int c = 0; c < 64; ++c) {
        __builtin_nontemporal_store(val, (v4f*)base); // global_store_b128 nt
        base += IMG_FLT;
    }
}

// ---------------------------------------------------------------------------
extern "C" void kernel_launch(void* const* d_in, const int* in_sizes, int n_in,
                              void* d_out, int out_size, void* d_ws, size_t ws_size,
                              hipStream_t stream)
{
    // input order per setup_inputs():
    // 0 point_cloud (dead code)  1 view_features  2 text_features
    // 3 w1  4 b1  5 w2  6 b2  7 wt  8 wc  9 wclip  10 num_candidates  11 num_views
    const float* vf    = (const float*)d_in[1];
    const float* txt   = (const float*)d_in[2];
    const float* w1    = (const float*)d_in[3];
    const float* b1    = (const float*)d_in[4];
    const float* w2    = (const float*)d_in[5];
    const float* b2    = (const float*)d_in[6];
    const float* wt    = (const float*)d_in[7];
    const float* wc    = (const float*)d_in[8];
    const float* wclip = (const float*)d_in[9];

    float* out = (float*)d_out;
    float* tr  = (float*)d_ws;          // [1024] text relevance
    float* cov = tr + NCAND;            // [1024] coverage

    dvs_relevance_kernel<<<NCAND, 64, 0, stream>>>(vf, txt, tr);
    dvs_score_kernel<<<NCAND / 16, 128, 0, stream>>>(vf, w1, b1, w2, b2, cov);
    dvs_select_kernel<<<1, 1024, 0, stream>>>(vf, tr, cov, wt, wc, wclip, out);

    dim3 rgrid(IMG_FLT / 4 / 256, 16);  // (147, 16)
    dvs_render_kernel<<<rgrid, 256, 0, stream>>>(out + NVIEWS * DFEAT + NVIEWS);
}
